// Net_58729382805605
// MI455X (gfx1250) — compile-verified, matrix operand order checked by reading, not run on hardware
//
#include <hip/hip_runtime.h>
#include <math.h>

// ---------------- problem constants ----------------
#define NN      100000     // nodes
#define F_IN    500        // input features
#define KPAD    512        // padded K for bf16 WMMA (16 steps of 32)
#define HC1     64         // H1*C1
#define H1      8
#define C1      8
#define C2      7
#define NEG_SLOPE 0.2f

typedef __attribute__((ext_vector_type(16))) __bf16          v16bf;
typedef __attribute__((ext_vector_type(16))) unsigned short  v16us;
typedef __attribute__((ext_vector_type(8)))  float           v8f;

// float -> bf16 bits, round-to-nearest-even
__device__ __forceinline__ unsigned short f2bf(float f) {
    unsigned u = __float_as_uint(f);
    u += 0x7fffu + ((u >> 16) & 1u);
    return (unsigned short)(u >> 16);
}

// order-preserving encode/decode for float atomicMax on unsigned
__device__ __forceinline__ unsigned encf(float f) {
    unsigned u = __float_as_uint(f);
    return (u & 0x80000000u) ? ~u : (u | 0x80000000u);
}
__device__ __forceinline__ float decf(unsigned o) {
    unsigned u = (o & 0x80000000u) ? (o & 0x7fffffffu) : ~o;
    return __uint_as_float(u);
}
#define ENC_NEG_INF 0x007FFFFFu   // encf(-inf)

// ---------------- init scratch ----------------
__global__ void init_ws(float* out1, unsigned* m1, float* s1,
                        unsigned* m2, float* s2, float* out2, float* calib) {
    int i = blockIdx.x * blockDim.x + threadIdx.x;
    int total = NN * HC1;
    for (; i < total; i += gridDim.x * blockDim.x) {
        out1[i] = 0.f;
        if (i < NN * H1) { m1[i] = ENC_NEG_INF; s1[i] = 0.f; }
        if (i < NN)      { m2[i] = ENC_NEG_INF; s2[i] = 0.f; }
        if (i < NN * C2) { out2[i] = 0.f; }
        if (i == 0)      { calib[0] = 0.f; }
    }
}

// ---------------- W1 -> W1^T bf16, K padded to 512 ----------------
__global__ void cvt_w1t(const float* __restrict__ W1, unsigned short* __restrict__ w1t) {
    int i = blockIdx.x * blockDim.x + threadIdx.x;     // [64][512]
    if (i >= HC1 * KPAD) return;
    int c = i >> 9, k = i & (KPAD - 1);
    w1t[i] = (k < F_IN) ? f2bf(W1[k * HC1 + c]) : (unsigned short)0;
}

// ---------------- GEMM1: h1[N][64] = x[N][500] @ W1, via WMMA bf16 ----------------
// 1 block = 16 rows; 4 waves, wave w owns cols [16w,16w+16)
__global__ __launch_bounds__(128) void gemm1_wmma(const float* __restrict__ x,
                                                  const unsigned short* __restrict__ w1t,
                                                  float* __restrict__ h1) {
    const int lane    = threadIdx.x & 31;
    const int wv      = threadIdx.x >> 5;
    const int rowBase = blockIdx.x * 16;
    const int m       = lane & 15;
    const int koff    = (lane >> 4) * 16;      // A/B K sub-offset per half-wave
    const int row     = rowBase + m;

    const float*          arow = x   + (long)row * F_IN;
    const unsigned short* bcol = w1t + (wv * 16 + m) * KPAD;

    v8f acc = {};

    // 15 full K-steps (K = 0..479), all in-bounds since 448+31 < 500
    for (int kk = 0; kk < 480; kk += 32) {
        __builtin_prefetch(arow + kk + 64, 0, 1);           // global_prefetch_b8
        v16us ab;
        const float4* ap = (const float4*)(arow + kk + koff); // 16B aligned (500*4 % 16 == 0)
        #pragma unroll
        for (int q = 0; q < 4; ++q) {
            float4 f = ap[q];
            ab[q * 4 + 0] = f2bf(f.x);
            ab[q * 4 + 1] = f2bf(f.y);
            ab[q * 4 + 2] = f2bf(f.z);
            ab[q * 4 + 3] = f2bf(f.w);
        }
        v16bf a = __builtin_bit_cast(v16bf, ab);
        v16bf b = *(const v16bf*)(bcol + kk + koff);
        acc = __builtin_amdgcn_wmma_f32_16x16x32_bf16(false, a, false, b,
                                                      (short)0, acc, false, false);
    }
    // tail K-step (K = 480..511), branchless:
    //   koff==0  lanes cover K 480..495 : all valid
    //   koff==16 lanes cover K 496..511 : first 4 valid (496..499 = last float4 of row),
    //            remaining 12 read from a clamped in-bounds pointer and select-zeroed.
    {
        const bool  lo   = (koff == 0);
        const float4* ap0 = (const float4*)(arow + 480 + koff);        // 480 or 496: in-bounds
        const float4* apr = (const float4*)(arow + (lo ? 484 : 480));  // clamped, in-bounds
        float4 f0 = ap0[0];
        float4 f1 = apr[0], f2 = apr[1], f3 = apr[2];
        v16us ab;
        ab[0] = f2bf(f0.x); ab[1] = f2bf(f0.y); ab[2] = f2bf(f0.z); ab[3] = f2bf(f0.w);
        ab[4]  = lo ? f2bf(f1.x) : (unsigned short)0;
        ab[5]  = lo ? f2bf(f1.y) : (unsigned short)0;
        ab[6]  = lo ? f2bf(f1.z) : (unsigned short)0;
        ab[7]  = lo ? f2bf(f1.w) : (unsigned short)0;
        ab[8]  = lo ? f2bf(f2.x) : (unsigned short)0;
        ab[9]  = lo ? f2bf(f2.y) : (unsigned short)0;
        ab[10] = lo ? f2bf(f2.z) : (unsigned short)0;
        ab[11] = lo ? f2bf(f2.w) : (unsigned short)0;
        ab[12] = lo ? f2bf(f3.x) : (unsigned short)0;
        ab[13] = lo ? f2bf(f3.y) : (unsigned short)0;
        ab[14] = lo ? f2bf(f3.z) : (unsigned short)0;
        ab[15] = lo ? f2bf(f3.w) : (unsigned short)0;
        v16bf a = __builtin_bit_cast(v16bf, ab);
        v16bf b = *(const v16bf*)(bcol + 480 + koff);   // B zero-padded beyond K=500
        acc = __builtin_amdgcn_wmma_f32_16x16x32_bf16(false, a, false, b,
                                                      (short)0, acc, false, false);
    }

    // C/D layout: lanes 0-15 rows r, lanes 16-31 rows r+8; col = lane&15
    const int col  = wv * 16 + m;
    const int rsel = (lane >> 4) * 8;
    #pragma unroll
    for (int r = 0; r < 8; ++r)
        h1[(long)(rowBase + rsel + r) * HC1 + col] = acc[r];
}

// ---------------- attention coefficients, layer 1 ----------------
__global__ void attn1(const float* __restrict__ h1,
                      const float* __restrict__ a_src, const float* __restrict__ a_dst,
                      float* __restrict__ asrc, float* __restrict__ adst) {
    int i = blockIdx.x * blockDim.x + threadIdx.x;   // n*8 + h
    if (i >= NN * H1) return;
    int n = i >> 3, h = i & 7;
    const float* hp = h1 + (long)n * HC1 + h * C1;
    float s = 0.f, d = 0.f;
    #pragma unroll
    for (int c = 0; c < C1; ++c) {
        float v = hp[c];
        s += v * a_src[h * C1 + c];
        d += v * a_dst[h * C1 + c];
    }
    asrc[i] = s; adst[i] = d;
}

// ---------------- layer-1 edge passes ----------------
__device__ __forceinline__ void edge_sd(const int* ei, int E, int idx, int& s, int& d) {
    if (idx < E) { s = ei[idx]; d = ei[E + idx]; }
    else         { s = d = idx - E; }
}

__global__ void edge_max1(const int* __restrict__ ei, int E, int ET,
                          const float* __restrict__ asrc, const float* __restrict__ adst,
                          unsigned* __restrict__ m1) {
    int idx = blockIdx.x * blockDim.x + threadIdx.x;
    if (idx >= ET) return;
    int s, d; edge_sd(ei, E, idx, s, d);
    #pragma unroll
    for (int h = 0; h < H1; ++h) {
        float e = asrc[s * H1 + h] + adst[d * H1 + h];
        e = e > 0.f ? e : NEG_SLOPE * e;
        atomicMax(m1 + d * H1 + h, encf(e));
    }
}

__global__ void edge_sum1(const int* __restrict__ ei, int E, int ET,
                          const float* __restrict__ asrc, const float* __restrict__ adst,
                          const unsigned* __restrict__ m1, float* __restrict__ s1) {
    int idx = blockIdx.x * blockDim.x + threadIdx.x;
    if (idx >= ET) return;
    int s, d; edge_sd(ei, E, idx, s, d);
    #pragma unroll
    for (int h = 0; h < H1; ++h) {
        float e = asrc[s * H1 + h] + adst[d * H1 + h];
        e = e > 0.f ? e : NEG_SLOPE * e;
        atomicAdd(s1 + d * H1 + h, __expf(e - decf(m1[d * H1 + h])));
    }
}

__global__ void edge_acc1(const int* __restrict__ ei, int E, int ET,
                          const float* __restrict__ asrc, const float* __restrict__ adst,
                          const unsigned* __restrict__ m1, const float* __restrict__ s1,
                          const float* __restrict__ h1, float* __restrict__ out1) {
    int idx = blockIdx.x * blockDim.x + threadIdx.x;   // (edge, head)
    if (idx >= ET * H1) return;
    int eidx = idx >> 3, h = idx & 7;
    int s, d; edge_sd(ei, E, eidx, s, d);
    float e = asrc[s * H1 + h] + adst[d * H1 + h];
    e = e > 0.f ? e : NEG_SLOPE * e;
    float w = __expf(e - decf(m1[d * H1 + h])) / (s1[d * H1 + h] + 1e-16f);
    const float* hp = h1 + (long)s * HC1 + h * C1;
    float*       op = out1 + (long)d * HC1 + h * C1;
    #pragma unroll
    for (int c = 0; c < C1; ++c)
        atomicAdd(op + c, hp[c] * w);
}

// ---------------- bias + relu + dropout ----------------
__global__ void act1(const float* __restrict__ out1, const float* __restrict__ b1,
                     const float* __restrict__ drop, float* __restrict__ h2in) {
    int i = blockIdx.x * blockDim.x + threadIdx.x;
    if (i >= NN * HC1) return;
    float v = out1[i] + b1[i & (HC1 - 1)];
    v = v > 0.f ? v : 0.f;
    h2in[i] = v * drop[i];
}

// ---------------- layer-2 linear (64 -> 7), scalar FMA ----------------
__global__ void gemm2(const float* __restrict__ h2in, const float* __restrict__ W2,
                      float* __restrict__ olin) {
    int i = blockIdx.x * blockDim.x + threadIdx.x;   // n*7 + c
    if (i >= NN * C2) return;
    int n = i / C2, c = i % C2;
    const float* hp = h2in + (long)n * HC1;
    float acc = 0.f;
    #pragma unroll
    for (int k = 0; k < HC1; ++k)
        acc = fmaf(hp[k], W2[k * C2 + c], acc);
    olin[i] = acc;
}

__global__ void attn2(const float* __restrict__ olin,
                      const float* __restrict__ a_src, const float* __restrict__ a_dst,
                      float* __restrict__ asrc, float* __restrict__ adst) {
    int n = blockIdx.x * blockDim.x + threadIdx.x;
    if (n >= NN) return;
    const float* op = olin + (long)n * C2;
    float s = 0.f, d = 0.f;
    #pragma unroll
    for (int c = 0; c < C2; ++c) { s += op[c] * a_src[c]; d += op[c] * a_dst[c]; }
    asrc[n] = s; adst[n] = d;
}

// ---------------- layer-2 edge passes (H=1) ----------------
__global__ void edge_max2(const int* __restrict__ ei, int E, int ET,
                          const float* __restrict__ asrc, const float* __restrict__ adst,
                          unsigned* __restrict__ m2) {
    int idx = blockIdx.x * blockDim.x + threadIdx.x;
    if (idx >= ET) return;
    int s, d; edge_sd(ei, E, idx, s, d);
    float e = asrc[s] + adst[d];
    e = e > 0.f ? e : NEG_SLOPE * e;
    atomicMax(m2 + d, encf(e));
}

__global__ void edge_sum2(const int* __restrict__ ei, int E, int ET,
                          const float* __restrict__ asrc, const float* __restrict__ adst,
                          const unsigned* __restrict__ m2, float* __restrict__ s2) {
    int idx = blockIdx.x * blockDim.x + threadIdx.x;
    if (idx >= ET) return;
    int s, d; edge_sd(ei, E, idx, s, d);
    float e = asrc[s] + adst[d];
    e = e > 0.f ? e : NEG_SLOPE * e;
    atomicAdd(s2 + d, __expf(e - decf(m2[d])));
}

__global__ void edge_acc2(const int* __restrict__ ei, int E, int ET,
                          const float* __restrict__ asrc, const float* __restrict__ adst,
                          const unsigned* __restrict__ m2, const float* __restrict__ s2,
                          const float* __restrict__ olin, float* __restrict__ out2) {
    int idx = blockIdx.x * blockDim.x + threadIdx.x;
    if (idx >= ET) return;
    int s, d; edge_sd(ei, E, idx, s, d);
    float e = asrc[s] + adst[d];
    e = e > 0.f ? e : NEG_SLOPE * e;
    float w = __expf(e - decf(m2[d])) / (s2[d] + 1e-16f);
    const float* op = olin + (long)s * C2;
    float*       dp = out2 + (long)d * C2;
    #pragma unroll
    for (int c = 0; c < C2; ++c)
        atomicAdd(dp + c, op[c] * w);
}

// ---------------- final: softmax / log-softmax / calib ----------------
__global__ void final_k(const float* __restrict__ out2, const float* __restrict__ b2,
                        float* __restrict__ dout, float* __restrict__ calib) {
    int n = blockIdx.x * blockDim.x + threadIdx.x;
    if (n >= NN) return;
    float v[C2], p[C2];
    float mx = -1e30f;
    #pragma unroll
    for (int c = 0; c < C2; ++c) { v[c] = out2[n * C2 + c] + b2[c]; mx = fmaxf(mx, v[c]); }
    float se = 0.f;
    #pragma unroll
    for (int c = 0; c < C2; ++c) { p[c] = __expf(v[c] - mx); se += p[c]; }
    float inv = 1.f / se, lse = logf(se);
    float t1 = -1.f, t2 = -1.f;
    #pragma unroll
    for (int c = 0; c < C2; ++c) {
        float pc = p[c] * inv; p[c] = pc;
        if (pc > t1) { t2 = t1; t1 = pc; } else if (pc > t2) { t2 = pc; }
    }
    #pragma unroll
    for (int c = 0; c < C2; ++c) {
        dout[(long)n * C2 + c] = (v[c] - mx) - lse;             // log_softmax
        dout[(long)NN * C2 + 1 + (long)n * C2 + c] = p[c];      // softmax
    }
    atomicAdd(calib, 1.f - t1 + t2);
}

__global__ void finalize_calib(const float* __restrict__ calib, float* __restrict__ dout) {
    if (threadIdx.x == 0 && blockIdx.x == 0)
        dout[(long)NN * C2] = calib[0] / (float)NN;
}

// ---------------- host launcher ----------------
extern "C" void kernel_launch(void* const* d_in, const int* in_sizes, int n_in,
                              void* d_out, int out_size, void* d_ws, size_t ws_size,
                              hipStream_t stream) {
    const float* x     = (const float*)d_in[0];
    const int*   ei    = (const int*)  d_in[1];
    const float* drop  = (const float*)d_in[2];
    const float* W1    = (const float*)d_in[3];
    const float* as1   = (const float*)d_in[4];
    const float* ad1   = (const float*)d_in[5];
    const float* b1    = (const float*)d_in[6];
    const float* W2    = (const float*)d_in[7];
    const float* as2   = (const float*)d_in[8];
    const float* ad2   = (const float*)d_in[9];
    const float* b2    = (const float*)d_in[10];
    float* dout = (float*)d_out;

    const int E  = in_sizes[1] / 2;
    const int ET = E + NN;

    // ---- carve workspace ----
    char* ws = (char*)d_ws;
    size_t off = 0;
    auto carve = [&](size_t bytes) { char* p = ws + off; off = (off + bytes + 255) & ~(size_t)255; return p; };
    unsigned short* w1t  = (unsigned short*)carve((size_t)HC1 * KPAD * 2);
    float*    h1    = (float*)   carve((size_t)NN * HC1 * 4);
    float*    asrc1 = (float*)   carve((size_t)NN * H1 * 4);
    float*    adst1 = (float*)   carve((size_t)NN * H1 * 4);
    unsigned* m1    = (unsigned*)carve((size_t)NN * H1 * 4);
    float*    s1    = (float*)   carve((size_t)NN * H1 * 4);
    float*    out1  = (float*)   carve((size_t)NN * HC1 * 4);
    float*    h2in  = (float*)   carve((size_t)NN * HC1 * 4);
    float*    olin  = (float*)   carve((size_t)NN * C2 * 4);
    float*    asrc2 = (float*)   carve((size_t)NN * 4);
    float*    adst2 = (float*)   carve((size_t)NN * 4);
    unsigned* m2    = (unsigned*)carve((size_t)NN * 4);
    float*    s2    = (float*)   carve((size_t)NN * 4);
    float*    out2  = (float*)   carve((size_t)NN * C2 * 4);
    float*    calib = (float*)   carve(256);
    (void)ws_size; (void)n_in; (void)out_size;

    const int T = 256;
    // init
    init_ws<<<4096, T, 0, stream>>>(out1, m1, s1, m2, s2, out2, calib);
    // W1 transpose+convert
    cvt_w1t<<<(HC1 * KPAD + T - 1) / T, T, 0, stream>>>(W1, w1t);
    // GEMM1 (WMMA bf16)
    gemm1_wmma<<<NN / 16, 128, 0, stream>>>(x, w1t, h1);
    // attention coefficients
    attn1<<<(NN * H1 + T - 1) / T, T, 0, stream>>>(h1, as1, ad1, asrc1, adst1);
    // layer-1 edge passes
    edge_max1<<<(ET + T - 1) / T, T, 0, stream>>>(ei, E, ET, asrc1, adst1, m1);
    edge_sum1<<<(ET + T - 1) / T, T, 0, stream>>>(ei, E, ET, asrc1, adst1, m1, s1);
    edge_acc1<<<(ET * H1 + T - 1) / T, T, 0, stream>>>(ei, E, ET, asrc1, adst1, m1, s1, h1, out1);
    // activation + dropout
    act1<<<(NN * HC1 + T - 1) / T, T, 0, stream>>>(out1, b1, drop, h2in);
    // layer-2 linear + attention
    gemm2<<<(NN * C2 + T - 1) / T, T, 0, stream>>>(h2in, W2, olin);
    attn2<<<(NN + T - 1) / T, T, 0, stream>>>(olin, as2, ad2, asrc2, adst2);
    // layer-2 edge passes
    edge_max2<<<(ET + T - 1) / T, T, 0, stream>>>(ei, E, ET, asrc2, adst2, m2);
    edge_sum2<<<(ET + T - 1) / T, T, 0, stream>>>(ei, E, ET, asrc2, adst2, m2, s2);
    edge_acc2<<<(ET + T - 1) / T, T, 0, stream>>>(ei, E, ET, asrc2, adst2, m2, s2, olin, out2);
    // outputs
    final_k<<<(NN + T - 1) / T, T, 0, stream>>>(out2, b2, dout, calib);
    finalize_calib<<<1, 64, 0, stream>>>(calib, dout);
}
